// SelfAttention_91293824844272
// MI455X (gfx1250) — matrix operand order repaired and checked by reading, not compiled, not run on hardware
//
#include <hip/hip_runtime.h>
#include <hip/hip_bf16.h>
#include <stdint.h>

#define NB     8
#define SEQL   2048
#define NPTS   (NB * SEQL)        // 16384
#define DIMX   256
#define HEADS  8
#define HDIM   32
#define KNN    16
#define NE     (NPTS * KNN)       // 262144
#define QKVD   (3 * DIMX)         // 768
#define KDIM   256                // contraction dim of BOTH GEMMs (compile-time)

static constexpr float ATTN_SCALE = 0.17677669529663687f; // 32^-0.5

typedef __attribute__((ext_vector_type(16))) __bf16 v16bf;
typedef __attribute__((ext_vector_type(8)))  float  v8f;

struct BF16x16 { uint4 lo, hi; };

__device__ __forceinline__ v16bf load_frag(const __hip_bfloat16* p0,
                                           const __hip_bfloat16* p1) {
  BF16x16 t;
  t.lo = *reinterpret_cast<const uint4*>(p0);
  t.hi = *reinterpret_cast<const uint4*>(p1);
  return __builtin_bit_cast(v16bf, t);
}

// ---------------------------------------------------------------------------
// fp32 -> bf16 conversion (grid-stride)
// ---------------------------------------------------------------------------
__global__ void k_f32_to_bf16(const float* __restrict__ src,
                              __hip_bfloat16* __restrict__ dst, int n) {
  for (int i = blockIdx.x * blockDim.x + threadIdx.x; i < n;
       i += gridDim.x * blockDim.x)
    dst[i] = __float2bfloat16(src[i]);
}

// ---------------------------------------------------------------------------
// C[M x Nn] = A[M x KDIM](bf16) * W[Nn x KDIM]^T(bf16) (+ bias), fp32 out.
// One wave -> 32(M) x 64(N) tile: per K-step, 4 shared B fragments feed
// 8x v_wmma_f32_16x16x32_bf16 (2 M-halves). K-loop fully unrolled so the
// scheduler can overlap fragment loads with WMMAs (partial loadcnt waits).
// Grids sized exactly: EXEC is all-ones for every WMMA (ISA requirement).
// ---------------------------------------------------------------------------
__global__ void k_gemm_bf16_wmma(const __hip_bfloat16* __restrict__ A,
                                 const __hip_bfloat16* __restrict__ W,
                                 float* __restrict__ C,
                                 const float* __restrict__ bias,
                                 int M, int Nn) {
  const int wid  = (blockIdx.x * blockDim.x + threadIdx.x) >> 5;
  const int lane = threadIdx.x & 31;
  const int lrow = lane & 15;
  const int lhi  = lane >> 4;
  const int ntn  = Nn >> 6;                 // # of 64-wide N tiles
  const int m0   = (wid / ntn) << 5;        // 32 M-rows per wave
  const int n0   = (wid % ntn) << 6;
  (void)M;

  v8f acc[2][4] = {};
  const __hip_bfloat16* arow0 = A + (size_t)(m0 + lrow) * KDIM;
  const __hip_bfloat16* arow1 = arow0 + (size_t)16 * KDIM;
  const __hip_bfloat16* wbase = W + (size_t)(n0 + lrow) * KDIM + 8 * lhi;

#pragma unroll
  for (int k0 = 0; k0 < KDIM; k0 += 32) {
    // A 16x32 fragments: halves 0..7 -> K=k0+8*lhi..+7, 8..15 -> K=k0+16+8*lhi..+7
    v16bf a0 = load_frag(arow0 + k0 + 8 * lhi, arow0 + k0 + 16 + 8 * lhi);
    v16bf a1 = load_frag(arow1 + k0 + 8 * lhi, arow1 + k0 + 16 + 8 * lhi);
    // B 32x16 fragments: lane -> N, halves 0..7 -> K=k0+8*lhi..+7, 8..15 -> K=k0+16+8*lhi..+7
    v16bf b0, b1, b2, b3;
    {
      const __hip_bfloat16* bp0 = wbase + k0;
      const __hip_bfloat16* bp1 = bp0 + (size_t)16 * KDIM;
      const __hip_bfloat16* bp2 = bp0 + (size_t)32 * KDIM;
      const __hip_bfloat16* bp3 = bp0 + (size_t)48 * KDIM;
      b0 = load_frag(bp0, bp0 + 16);
      b1 = load_frag(bp1, bp1 + 16);
      b2 = load_frag(bp2, bp2 + 16);
      b3 = load_frag(bp3, bp3 + 16);
    }
    acc[0][0] = __builtin_amdgcn_wmma_f32_16x16x32_bf16(false, a0, false, b0, (short)0, acc[0][0], false, false);
    acc[1][0] = __builtin_amdgcn_wmma_f32_16x16x32_bf16(false, a1, false, b0, (short)0, acc[1][0], false, false);
    acc[0][1] = __builtin_amdgcn_wmma_f32_16x16x32_bf16(false, a0, false, b1, (short)0, acc[0][1], false, false);
    acc[1][1] = __builtin_amdgcn_wmma_f32_16x16x32_bf16(false, a1, false, b1, (short)0, acc[1][1], false, false);
    acc[0][2] = __builtin_amdgcn_wmma_f32_16x16x32_bf16(false, a0, false, b2, (short)0, acc[0][2], false, false);
    acc[1][2] = __builtin_amdgcn_wmma_f32_16x16x32_bf16(false, a1, false, b2, (short)0, acc[1][2], false, false);
    acc[0][3] = __builtin_amdgcn_wmma_f32_16x16x32_bf16(false, a0, false, b3, (short)0, acc[0][3], false, false);
    acc[1][3] = __builtin_amdgcn_wmma_f32_16x16x32_bf16(false, a1, false, b3, (short)0, acc[1][3], false, false);
  }

#pragma unroll
  for (int s = 0; s < 4; ++s) {
    const int cn = n0 + s * 16 + lrow;
    const float bv = bias ? bias[cn] : 0.0f;
#pragma unroll
    for (int half = 0; half < 2; ++half) {
#pragma unroll
      for (int r = 0; r < 8; ++r) {
        const int rm = m0 + half * 16 + r + 8 * lhi;  // C/D: VGPR r -> M=r+8*lhi
        C[(size_t)rm * Nn + cn] = acc[half][s][r] + bv;
      }
    }
  }
}

// ---------------------------------------------------------------------------
// kNN (K=16, no self-loop) within each contiguous batch of 2048 points.
// One thread per query row; candidates staged through LDS in tiles of 256.
// Register-resident sorted top-16, threshold-guarded insertion (rare path).
// ---------------------------------------------------------------------------
__global__ void k_knn(const float* __restrict__ xyz, int* __restrict__ col) {
  __shared__ float sx[256], sy[256], sz[256];
  const int b      = blockIdx.x >> 3;
  const int chunk  = blockIdx.x & 7;
  const int tid    = threadIdx.x;
  const int rlocal = chunk * 256 + tid;
  const int base   = b * SEQL;

  const float px = xyz[(size_t)(base + rlocal) * 3 + 0];
  const float py = xyz[(size_t)(base + rlocal) * 3 + 1];
  const float pz = xyz[(size_t)(base + rlocal) * 3 + 2];

  float bd[KNN];
  int   bi[KNN];
#pragma unroll
  for (int j = 0; j < KNN; ++j) { bd[j] = 3.0e38f; bi[j] = 0; }

  for (int t = 0; t < SEQL / 256; ++t) {
    const int cg = base + t * 256 + tid;
    sx[tid] = xyz[(size_t)cg * 3 + 0];
    sy[tid] = xyz[(size_t)cg * 3 + 1];
    sz[tid] = xyz[(size_t)cg * 3 + 2];
    __syncthreads();
    for (int c = 0; c < 256; ++c) {
      const int cl = t * 256 + c;
      const float dx = px - sx[c];
      const float dy = py - sy[c];
      const float dz = pz - sz[c];
      const float d2 = fmaf(dx, dx, fmaf(dy, dy, dz * dz));
      if (cl != rlocal && d2 < bd[KNN - 1]) {
        bd[KNN - 1] = d2;
        bi[KNN - 1] = cl;
#pragma unroll
        for (int j = KNN - 1; j > 0; --j) {
          if (bd[j] < bd[j - 1]) {
            float td = bd[j]; bd[j] = bd[j - 1]; bd[j - 1] = td;
            int   ti = bi[j]; bi[j] = bi[j - 1]; bi[j - 1] = ti;
          }
        }
      }
    }
    __syncthreads();
  }
#pragma unroll
  for (int j = 0; j < KNN; ++j)
    col[(size_t)(base + rlocal) * KNN + j] = base + bi[j];
}

// ---------------------------------------------------------------------------
// Edge logits: logits[e][h] = scale * dot(q[row(e)][h], k[col(e)][h])
// ---------------------------------------------------------------------------
__global__ void k_edge_logits(const float* __restrict__ qkv,
                              const int* __restrict__ col,
                              float* __restrict__ logits) {
  const int gid = blockIdx.x * blockDim.x + threadIdx.x; // NE*HEADS threads
  const int e = gid >> 3;
  const int h = gid & 7;
  const int r = e >> 4;           // row = repeat(arange(N), K)
  const int c = col[e];
  const float4* q = reinterpret_cast<const float4*>(qkv + (size_t)r * QKVD + h * HDIM);
  const float4* k = reinterpret_cast<const float4*>(qkv + (size_t)c * QKVD + DIMX + h * HDIM);
  float acc = 0.0f;
#pragma unroll
  for (int i = 0; i < HDIM / 4; ++i) {
    const float4 qa = q[i], ka = k[i];
    acc += qa.x * ka.x + qa.y * ka.y + qa.z * ka.z + qa.w * ka.w;
  }
  logits[(size_t)e * HEADS + h] = acc * ATTN_SCALE;
}

// ---------------------------------------------------------------------------
// Global (axis=0) softmax reductions per head — deterministic LDS trees.
// ---------------------------------------------------------------------------
__global__ void k_max_part(const float* __restrict__ logits,
                           float* __restrict__ part) {
  __shared__ float sm[HEADS * 256];
  const int tid = threadIdx.x;
  float m[HEADS];
#pragma unroll
  for (int h = 0; h < HEADS; ++h) m[h] = -3.0e38f;
  for (int e = blockIdx.x * 256 + tid; e < NE; e += 256 * 256) {
#pragma unroll
    for (int h = 0; h < HEADS; ++h)
      m[h] = fmaxf(m[h], logits[(size_t)e * HEADS + h]);
  }
#pragma unroll
  for (int h = 0; h < HEADS; ++h) sm[h * 256 + tid] = m[h];
  __syncthreads();
  for (int s = 128; s > 0; s >>= 1) {
    if (tid < s) {
#pragma unroll
      for (int h = 0; h < HEADS; ++h)
        sm[h * 256 + tid] = fmaxf(sm[h * 256 + tid], sm[h * 256 + tid + s]);
    }
    __syncthreads();
  }
  if (tid < HEADS) part[blockIdx.x * HEADS + tid] = sm[tid * 256];
}

__global__ void k_max_final(const float* __restrict__ part,
                            float* __restrict__ stats) {
  __shared__ float sm[HEADS * 256];
  const int tid = threadIdx.x;
#pragma unroll
  for (int h = 0; h < HEADS; ++h) sm[h * 256 + tid] = part[tid * HEADS + h];
  __syncthreads();
  for (int s = 128; s > 0; s >>= 1) {
    if (tid < s) {
#pragma unroll
      for (int h = 0; h < HEADS; ++h)
        sm[h * 256 + tid] = fmaxf(sm[h * 256 + tid], sm[h * 256 + tid + s]);
    }
    __syncthreads();
  }
  if (tid < HEADS) stats[tid] = sm[tid * 256];
}

__global__ void k_sum_part(const float* __restrict__ logits,
                           const float* __restrict__ stats,
                           float* __restrict__ part) {
  __shared__ float sm[HEADS * 256];
  const int tid = threadIdx.x;
  float mx[HEADS], s0[HEADS];
#pragma unroll
  for (int h = 0; h < HEADS; ++h) { mx[h] = stats[h]; s0[h] = 0.0f; }
  for (int e = blockIdx.x * 256 + tid; e < NE; e += 256 * 256) {
#pragma unroll
    for (int h = 0; h < HEADS; ++h)
      s0[h] += __expf(logits[(size_t)e * HEADS + h] - mx[h]);
  }
#pragma unroll
  for (int h = 0; h < HEADS; ++h) sm[h * 256 + tid] = s0[h];
  __syncthreads();
  for (int s = 128; s > 0; s >>= 1) {
    if (tid < s) {
#pragma unroll
      for (int h = 0; h < HEADS; ++h)
        sm[h * 256 + tid] += sm[h * 256 + tid + s];
    }
    __syncthreads();
  }
  if (tid < HEADS) part[blockIdx.x * HEADS + tid] = sm[tid * 256];
}

__global__ void k_sum_final(const float* __restrict__ part,
                            float* __restrict__ stats) {
  __shared__ float sm[HEADS * 256];
  const int tid = threadIdx.x;
#pragma unroll
  for (int h = 0; h < HEADS; ++h) sm[h * 256 + tid] = part[tid * HEADS + h];
  __syncthreads();
  for (int s = 128; s > 0; s >>= 1) {
    if (tid < s) {
#pragma unroll
      for (int h = 0; h < HEADS; ++h)
        sm[h * 256 + tid] += sm[h * 256 + tid + s];
    }
    __syncthreads();
  }
  if (tid < HEADS) stats[HEADS + tid] = 1.0f / sm[tid * 256];
}

// ---------------------------------------------------------------------------
// Weighted aggregation: agg[n][d] = sum_j softmax_w(n*K+j, h=d/32) * v[col][d]
// (rows are exactly K=16 contiguous, sorted edges -> no atomics needed)
// ---------------------------------------------------------------------------
__global__ void k_aggregate(const float* __restrict__ qkv,
                            const int* __restrict__ col,
                            const float* __restrict__ logits,
                            const float* __restrict__ stats,
                            float* __restrict__ agg) {
  const int gid = blockIdx.x * blockDim.x + threadIdx.x; // NPTS*DIMX threads
  const int n = gid >> 8;
  const int d = gid & 255;
  const int h = d >> 5;
  const float mx  = stats[h];
  const float inv = stats[HEADS + h];
  float acc = 0.0f;
#pragma unroll
  for (int j = 0; j < KNN; ++j) {
    const int e = n * KNN + j;
    const int c = col[e];
    const float w = __expf(logits[(size_t)e * HEADS + h] - mx) * inv;
    acc = fmaf(w, qkv[(size_t)c * QKVD + 2 * DIMX + d], acc);
  }
  agg[(size_t)n * DIMX + d] = acc;
}

// ---------------------------------------------------------------------------
extern "C" void kernel_launch(void* const* d_in, const int* in_sizes, int n_in,
                              void* d_out, int out_size, void* d_ws, size_t ws_size,
                              hipStream_t stream) {
  (void)in_sizes; (void)n_in; (void)out_size; (void)ws_size;
  const float* x     = (const float*)d_in[0];   // [N, 256]
  const float* xyz   = (const float*)d_in[1];   // [N, 3]
  // d_in[2] = batch indices (implicit: contiguous equal batches)
  const float* Wqkv  = (const float*)d_in[3];   // [768, 256]
  const float* Wproj = (const float*)d_in[4];   // [256, 256]
  const float* bproj = (const float*)d_in[5];   // [256]
  float* out = (float*)d_out;                   // [N, 256]

  char* ws = (char*)d_ws;
  size_t off = 0;
  auto take = [&](size_t bytes) -> char* {
    char* p = ws + off;
    off = (off + bytes + 255) & ~(size_t)255;
    return p;
  };
  __hip_bfloat16* xbf    = (__hip_bfloat16*)take((size_t)NPTS * DIMX * 2);
  __hip_bfloat16* wqkvbf = (__hip_bfloat16*)take((size_t)QKVD * DIMX * 2);
  __hip_bfloat16* wprjbf = (__hip_bfloat16*)take((size_t)DIMX * DIMX * 2);
  float*          qkv    = (float*)take((size_t)NPTS * QKVD * 4);
  int*            col    = (int*)take((size_t)NE * 4);
  float*          logits = (float*)take((size_t)NE * HEADS * 4);
  float*          part   = (float*)take((size_t)256 * HEADS * 4);
  float*          stats  = (float*)take((size_t)2 * HEADS * 4);
  float*          agg    = (float*)take((size_t)NPTS * DIMX * 4);
  __hip_bfloat16* aggbf  = (__hip_bfloat16*)take((size_t)NPTS * DIMX * 2);

  // 1) bf16 conversions for WMMA operands
  k_f32_to_bf16<<<4096, 256, 0, stream>>>(x, xbf, NPTS * DIMX);
  k_f32_to_bf16<<<(QKVD * DIMX + 255) / 256, 256, 0, stream>>>(Wqkv, wqkvbf, QKVD * DIMX);
  k_f32_to_bf16<<<(DIMX * DIMX + 255) / 256, 256, 0, stream>>>(Wproj, wprjbf, DIMX * DIMX);

  // 2) QKV projection: [16384,256] x [768,256]^T -> [16384,768]
  {
    const int waves = (NPTS / 32) * (QKVD / 64);   // 6144
    k_gemm_bf16_wmma<<<waves / 8, 256, 0, stream>>>(xbf, wqkvbf, qkv, nullptr,
                                                    NPTS, QKVD);
  }

  // 3) kNN edges
  k_knn<<<NB * (SEQL / 256), 256, 0, stream>>>(xyz, col);

  // 4) edge attention logits
  k_edge_logits<<<NE * HEADS / 256, 256, 0, stream>>>(qkv, col, logits);

  // 5) global per-head softmax stats (deterministic tree reductions)
  k_max_part <<<256, 256, 0, stream>>>(logits, part);
  k_max_final<<<1,   256, 0, stream>>>(part, stats);
  k_sum_part <<<256, 256, 0, stream>>>(logits, stats, part);
  k_sum_final<<<1,   256, 0, stream>>>(part, stats);

  // 6) weighted neighbor aggregation
  k_aggregate<<<NPTS * DIMX / 256, 256, 0, stream>>>(qkv, col, logits, stats, agg);

  // 7) output projection: [16384,256] x [256,256]^T + bias -> d_out
  k_f32_to_bf16<<<4096, 256, 0, stream>>>(agg, aggbf, NPTS * DIMX);
  {
    const int waves = (NPTS / 32) * (DIMX / 64);   // 2048
    k_gemm_bf16_wmma<<<waves / 8, 256, 0, stream>>>(aggbf, wprjbf, out, bproj,
                                                    NPTS, DIMX);
  }
}
